// Mamba2BCNetwork_9964324127540
// MI455X (gfx1250) — matrix operand order repaired and checked by reading, not compiled, not run on hardware
//
#include <hip/hip_runtime.h>
#include <hip/hip_bf16.h>
#include <math.h>

// ---- problem dimensions (fixed by reference) ----
#define D_MODEL     512
#define D_INNER     1024
#define D_STATE     16
#define D_CONV      4
#define DT_RANK     32
#define N_LAYERS    3
#define STATE_DIM   256
#define NUM_ACTIONS 9
#define BATCH       8
#define SEQ         1024
#define MROWS       (BATCH * SEQ)   // 8192 "token" rows

typedef __attribute__((ext_vector_type(16))) __bf16 v16bf;
typedef __attribute__((ext_vector_type(8)))  float  v8f;

#define ACT_NONE     0
#define ACT_RELU     1
#define ACT_SOFTPLUS 2

__device__ __forceinline__ float sigmoidf_(float x) { return 1.0f / (1.0f + __expf(-x)); }
__device__ __forceinline__ float siluf_(float x)    { return x * sigmoidf_(x); }
__device__ __forceinline__ float softplusf_(float x){ return (x > 20.0f) ? x : log1pf(__expf(x)); }

// ============================================================================
// WMMA GEMM: C[M,N] = act( A[M,K] * W[N,K]^T + bias[N] )
// A row-major with leading dim lda; W row-major N x K (contiguous K); C ldc.
// Block = WROWS*WCOLS waves (wave32); each wave computes a 32x32 tile via
// 2x2 v_wmma_f32_16x16x32_bf16. fp32 tiles converted to bf16 in LDS.
// MGUARD=false: M is an exact multiple of 32*WROWS -> no row guards, straight
// stores (avoids per-element exec-mask branches seen in round-1 disasm).
// Requires: K % 32 == 0, N % (32*WCOLS) == 0, lda/K multiples of 4.
// ============================================================================
template <int WROWS, int WCOLS, int ACT, bool BIAS, bool MGUARD>
__global__ __launch_bounds__(WROWS * WCOLS * 32)
void gemm_bf16_wmma(const float* __restrict__ A, const float* __restrict__ W,
                    const float* __restrict__ bias, float* __restrict__ C,
                    int M, int N, int K, int lda, int ldc) {
  constexpr int BM = 32 * WROWS;
  constexpr int BN = 32 * WCOLS;
  constexpr int KC = 32;
  constexpr int NT = WROWS * WCOLS * 32;

  __shared__ __bf16 As[BM][KC + 2];   // [row][k]   (+2 pad: bank skew)
  __shared__ __bf16 Bs[BN][KC + 2];   // [col][k]

  const int tid   = threadIdx.x;
  const int lane  = tid & 31;
  const int wid   = tid >> 5;
  const int wrow  = wid % WROWS;
  const int wcol  = wid / WROWS;
  const int lrow  = lane & 15;
  const int lhalf = lane >> 4;

  const int rowBase = blockIdx.x * BM;
  const int colBase = blockIdx.y * BN;

  v8f acc[2][2];
#pragma unroll
  for (int i = 0; i < 2; ++i)
#pragma unroll
    for (int j = 0; j < 2; ++j)
#pragma unroll
      for (int v = 0; v < 8; ++v) acc[i][j][v] = 0.0f;

  for (int kc = 0; kc < K; kc += KC) {
    // ---- stage A tile (fp32 -> bf16) ----
    for (int idx = tid * 4; idx < BM * KC; idx += NT * 4) {
      int r = idx >> 5, c = idx & 31;
      int gr = rowBase + r;
      float4 v4;
      if (MGUARD) {
        v4 = make_float4(0.f, 0.f, 0.f, 0.f);
        if (gr < M)
          v4 = *reinterpret_cast<const float4*>(&A[(long)gr * lda + kc + c]);
      } else {
        v4 = *reinterpret_cast<const float4*>(&A[(long)gr * lda + kc + c]);
      }
      As[r][c + 0] = (__bf16)v4.x; As[r][c + 1] = (__bf16)v4.y;
      As[r][c + 2] = (__bf16)v4.z; As[r][c + 3] = (__bf16)v4.w;
    }
    // ---- stage W tile as B^T: Bs[n][k] ----
    for (int idx = tid * 4; idx < BN * KC; idx += NT * 4) {
      int n = idx >> 5, c = idx & 31;
      int gn = colBase + n;
      float4 v4 = *reinterpret_cast<const float4*>(&W[(long)gn * K + kc + c]);
      Bs[n][c + 0] = (__bf16)v4.x; Bs[n][c + 1] = (__bf16)v4.y;
      Bs[n][c + 2] = (__bf16)v4.z; Bs[n][c + 3] = (__bf16)v4.w;
    }
    __syncthreads();

    // ---- build fragments per ISA 7.12.2 (wave32) ----
    v16bf afrag[2], bfrag[2];
#pragma unroll
    for (int i = 0; i < 2; ++i) {
      int r = wrow * 32 + i * 16 + lrow;
#pragma unroll
      for (int v = 0; v < 8; ++v) {
        // A 16x32 bf16: lanes0-15 -> K {0..7,16..23}; lanes16-31 -> K {8..15,24..31}
        int k0 = ((v < 4) ? (2 * v) : (16 + 2 * (v - 4))) + 8 * lhalf;
        afrag[i][2 * v]     = As[r][k0];
        afrag[i][2 * v + 1] = As[r][k0 + 1];
      }
    }
#pragma unroll
    for (int j = 0; j < 2; ++j) {
      int n = wcol * 32 + j * 16 + lrow;
#pragma unroll
      for (int v = 0; v < 8; ++v) {
        // B 32x16 bf16: lanes0-15 -> K {0..7,16..23}; lanes16-31 -> K {8..15,24..31}
        int k0 = ((v < 4) ? (2 * v) : (16 + 2 * (v - 4))) + 8 * lhalf;
        bfrag[j][2 * v]     = Bs[n][k0];
        bfrag[j][2 * v + 1] = Bs[n][k0 + 1];
      }
    }
#pragma unroll
    for (int i = 0; i < 2; ++i)
#pragma unroll
      for (int j = 0; j < 2; ++j)
        acc[i][j] = __builtin_amdgcn_wmma_f32_16x16x32_bf16(
            false, afrag[i], false, bfrag[j], (short)0, acc[i][j], false, false);
    __syncthreads();
  }

  // ---- epilogue: C/D layout -> M = v + 8*lhalf, N = lrow ----
#pragma unroll
  for (int i = 0; i < 2; ++i) {
#pragma unroll
    for (int j = 0; j < 2; ++j) {
      const int colT = colBase + wcol * 32 + j * 16 + lrow;
      float b = 0.0f;
      if (BIAS) b = bias[colT];
#pragma unroll
      for (int v = 0; v < 8; ++v) {
        int m   = v + 8 * lhalf;
        int row = rowBase + wrow * 32 + i * 16 + m;
        if (!MGUARD || row < M) {
          float val = acc[i][j][v];
          if (BIAS)                val += b;
          if (ACT == ACT_RELU)     val = fmaxf(val, 0.0f);
          if (ACT == ACT_SOFTPLUS) val = softplusf_(val);
          C[(long)row * ldc + colT] = val;
        }
      }
    }
  }
}

// ============================================================================
// Causal depthwise conv1d (width 4) + bias + SiLU.
// Reads xi half of XZ (stride 2048), writes XC (stride 1024).
// ============================================================================
__global__ __launch_bounds__(256)
void conv_silu_kernel(const float* __restrict__ XZ, const float* __restrict__ cw,
                      const float* __restrict__ cb, float* __restrict__ XC) {
  long idx = (long)blockIdx.x * blockDim.x + threadIdx.x;
  int  d   = (int)(idx & (D_INNER - 1));
  long row = idx >> 10;            // token row = b*SEQ + t
  int  t   = (int)(row & (SEQ - 1));
  long bL  = row - t;
  float acc = cb[d];
#pragma unroll
  for (int j = 0; j < D_CONV; ++j) {
    int tj = t - (D_CONV - 1) + j;
    if (tj >= 0)
      acc = fmaf(XZ[(bL + tj) * (2 * D_INNER) + d], cw[d * D_CONV + j], acc);
  }
  XC[idx] = siluf_(acc);
}

// ============================================================================
// Selective scan: lane-per-(b,d), 16-state recurrence in registers.
// B_t / C_t staged in LDS per 64-step chunk. Writes ys + xi*D.
// ============================================================================
#define SCAN_TT 64
#define SCAN_BD 128

__global__ __launch_bounds__(SCAN_BD)
void scan_kernel(const float* __restrict__ DTm, const float* __restrict__ XC,
                 const float* __restrict__ DBL, const float* __restrict__ A_log,
                 const float* __restrict__ Dvec, float* __restrict__ Y) {
  __shared__ float Bs[SCAN_TT][D_STATE];
  __shared__ float Cs[SCAN_TT][D_STATE];
  const int  b  = blockIdx.x;
  const int  d  = blockIdx.y * SCAN_BD + threadIdx.x;
  const long bL = (long)b * SEQ;

  float a[D_STATE], h[D_STATE];
#pragma unroll
  for (int s = 0; s < D_STATE; ++s) {
    a[s] = -__expf(A_log[d * D_STATE + s]);
    h[s] = 0.0f;
  }
  const float Dd = Dvec[d];

  for (int t0 = 0; t0 < SEQ; t0 += SCAN_TT) {
    if (t0 + SCAN_TT < SEQ) {  // gfx1250 global_prefetch of next chunk
      __builtin_prefetch(&DTm[(bL + t0 + SCAN_TT) * D_INNER + d], 0, 1);
      __builtin_prefetch(&XC [(bL + t0 + SCAN_TT) * D_INNER + d], 0, 1);
    }
    for (int i = threadIdx.x; i < SCAN_TT * D_STATE; i += SCAN_BD) {
      int tt = i >> 4, s = i & 15;
      const float* base = &DBL[(bL + t0 + tt) * (DT_RANK + 2 * D_STATE)];
      Bs[tt][s] = base[DT_RANK + s];
      Cs[tt][s] = base[DT_RANK + D_STATE + s];
    }
    __syncthreads();
    for (int tt = 0; tt < SCAN_TT; ++tt) {
      long  o   = (bL + t0 + tt) * D_INNER + d;
      float dt  = DTm[o];
      float x   = XC[o];
      float dtx = dt * x;
      float y   = x * Dd;
#pragma unroll
      for (int s = 0; s < D_STATE; ++s) {
        float dA = __expf(dt * a[s]);
        h[s] = fmaf(dA, h[s], dtx * Bs[tt][s]);
        y    = fmaf(h[s], Cs[tt][s], y);
      }
      Y[o] = y;
    }
    __syncthreads();
  }
}

// y *= silu(z)   (z = second half of XZ rows)
__global__ __launch_bounds__(256)
void gate_kernel(float* __restrict__ Y, const float* __restrict__ XZ) {
  long idx = (long)blockIdx.x * blockDim.x + threadIdx.x;
  int  d   = (int)(idx & (D_INNER - 1));
  long row = idx >> 10;
  float z  = XZ[row * (2 * D_INNER) + D_INNER + d];
  Y[idx] *= siluf_(z);
}

// logits = H(8x256) * head2_w(9x256)^T + b
__global__ __launch_bounds__(96)
void head2_kernel(const float* __restrict__ H, const float* __restrict__ W2,
                  const float* __restrict__ b2, float* __restrict__ out) {
  int tid = threadIdx.x;
  if (tid >= BATCH * NUM_ACTIONS) return;
  int b = tid / NUM_ACTIONS, a = tid % NUM_ACTIONS;
  float acc = b2[a];
  for (int h = 0; h < D_MODEL / 2; ++h)
    acc = fmaf(H[b * (D_MODEL / 2) + h], W2[a * (D_MODEL / 2) + h], acc);
  out[tid] = acc;
}

// ============================================================================
extern "C" void kernel_launch(void* const* d_in, const int* in_sizes, int n_in,
                              void* d_out, int out_size, void* d_ws, size_t ws_size,
                              hipStream_t stream) {
  const float* ss      = (const float*)d_in[0];
  const float* embed_w = (const float*)d_in[1];
  const float* embed_b = (const float*)d_in[2];
  const float* in_w    = (const float*)d_in[3];
  const float* conv_w  = (const float*)d_in[4];
  const float* conv_b  = (const float*)d_in[5];
  const float* x_w     = (const float*)d_in[6];
  const float* dt_w    = (const float*)d_in[7];
  const float* dt_b    = (const float*)d_in[8];
  const float* A_log   = (const float*)d_in[9];
  const float* Dv      = (const float*)d_in[10];
  const float* out_w   = (const float*)d_in[11];
  const float* head1_w = (const float*)d_in[12];
  const float* head1_b = (const float*)d_in[13];
  const float* head2_w = (const float*)d_in[14];
  const float* head2_b = (const float*)d_in[15];

  float* ws  = (float*)d_ws;
  float* X   = ws;                                   // 8192 x 512
  float* XZ  = X   + (long)MROWS * D_MODEL;          // 8192 x 2048
  float* XC  = XZ  + (long)MROWS * 2 * D_INNER;      // 8192 x 1024
  float* DBL = XC  + (long)MROWS * D_INNER;          // 8192 x 64
  float* DT  = DBL + (long)MROWS * (DT_RANK + 2 * D_STATE); // 8192 x 1024
  float* Y   = DT  + (long)MROWS * D_INNER;          // 8192 x 1024
  float* H   = Y   + (long)MROWS * D_INNER;          // 8 x 256

  const int M = MROWS;
  const long elemsPerTok = (long)M * D_INNER;        // 8,388,608 (% 256 == 0)

  // x = ss @ embed_w^T + embed_b
  gemm_bf16_wmma<2, 4, ACT_NONE, true, false>
      <<<dim3(M / 64, D_MODEL / 128), 256, 0, stream>>>(
          ss, embed_w, embed_b, X, M, D_MODEL, STATE_DIM, STATE_DIM, D_MODEL);

  for (int l = 0; l < N_LAYERS; ++l) {
    // xz = x @ in_proj_w^T
    gemm_bf16_wmma<2, 4, ACT_NONE, false, false>
        <<<dim3(M / 64, (2 * D_INNER) / 128), 256, 0, stream>>>(
            X, in_w + (long)l * 2 * D_INNER * D_MODEL, nullptr, XZ,
            M, 2 * D_INNER, D_MODEL, D_MODEL, 2 * D_INNER);
    // xi = silu(causal depthwise conv(xi) + cb)
    conv_silu_kernel<<<(unsigned)(elemsPerTok / 256), 256, 0, stream>>>(
        XZ, conv_w + (long)l * D_INNER * D_CONV, conv_b + (long)l * D_INNER, XC);
    // dbl = xi @ x_proj_w^T   (N = 64)
    gemm_bf16_wmma<2, 2, ACT_NONE, false, false>
        <<<dim3(M / 64, 1), 128, 0, stream>>>(
            XC, x_w + (long)l * (DT_RANK + 2 * D_STATE) * D_INNER, nullptr, DBL,
            M, DT_RANK + 2 * D_STATE, D_INNER, D_INNER, DT_RANK + 2 * D_STATE);
    // dt = softplus(dbl[:, :32] @ dt_proj_w^T + dt_b)
    gemm_bf16_wmma<2, 4, ACT_SOFTPLUS, true, false>
        <<<dim3(M / 64, D_INNER / 128), 256, 0, stream>>>(
            DBL, dt_w + (long)l * D_INNER * DT_RANK, dt_b + (long)l * D_INNER, DT,
            M, D_INNER, DT_RANK, DT_RANK + 2 * D_STATE, D_INNER);
    // selective scan (ys + xi*D)
    scan_kernel<<<dim3(BATCH, D_INNER / SCAN_BD), SCAN_BD, 0, stream>>>(
        DT, XC, DBL, A_log + (long)l * D_INNER * D_STATE, Dv + (long)l * D_INNER, Y);
    // y *= silu(z)
    gate_kernel<<<(unsigned)(elemsPerTok / 256), 256, 0, stream>>>(Y, XZ);
    // x = y @ out_proj_w^T
    gemm_bf16_wmma<2, 4, ACT_NONE, false, false>
        <<<dim3(M / 64, D_MODEL / 128), 256, 0, stream>>>(
            Y, out_w + (long)l * D_MODEL * D_INNER, nullptr, X,
            M, D_MODEL, D_INNER, D_INNER, D_MODEL);
  }

  // h = relu(x_last @ head1_w^T + b)   (M = 8, strided rows of X, needs guards)
  gemm_bf16_wmma<1, 2, ACT_RELU, true, true>
      <<<dim3(1, (D_MODEL / 2) / 64), 64, 0, stream>>>(
          X + (long)(SEQ - 1) * D_MODEL, head1_w, head1_b, H,
          BATCH, D_MODEL / 2, D_MODEL, SEQ * D_MODEL, D_MODEL / 2);
  // logits
  head2_kernel<<<1, 96, 0, stream>>>(H, head2_w, head2_b, (float*)d_out);
}
